// time_change_func_integral_32323923870367
// MI455X (gfx1250) — matrix operand order repaired and checked by reading, not compiled, not run on hardware
//
#include <hip/hip_runtime.h>
#include <hip/hip_bf16.h>
#include <cmath>

typedef __attribute__((ext_vector_type(16))) _Float16 v16h;
typedef __attribute__((ext_vector_type(8)))  float    v8f;

// Hardware-native stable softplus:
//   softplus(x) = max(x,0) + ln2 * log2(1 + 2^(-|x|*log2e))
// -> exactly one v_exp_f32 + one v_log_f32 + 5 simple VALU ops.
__device__ __forceinline__ float softplus_f(float x) {
  const float LOG2E = 1.4426950408889634f;
  const float LN2   = 0.6931471805599453f;
  float e = __builtin_amdgcn_exp2f(-fabsf(x) * LOG2E);  // v_exp_f32
  float l = __builtin_amdgcn_logf(1.0f + e);            // v_log_f32 (log2)
  return fmaxf(x, 0.0f) + l * LN2;
}

#define WAVES_PER_BLOCK 8

// ---------------------------------------------------------------------------
// Kernel 1: per-point MLP via f16 WMMA (16 points per wave-tile).
//
// A-operand (16x32 f16) layout per ISA 7.12.2: lane L holds row M=L&15;
//   element i -> K = ((i<8)?0:16) + (lane>=16 ? 8:0) + (i&7)
// B-operand (32x16 f16): lane L holds column N=(L&15); element i -> K = ((i<8)?0:16) + (lane>=16?8:0) + (i&7)
// C/D (16x16 f32): lane L holds N=(L&15); VGPR r -> M = r + (lane>=16 ? 8:0)
// ---------------------------------------------------------------------------
__global__ __launch_bounds__(256) void mlp_wmma_kernel(
    const float* __restrict__ t,
    const float* __restrict__ W1, const float* __restrict__ b1,
    const float* __restrict__ W2, const float* __restrict__ b2,
    const float* __restrict__ W3, const float* __restrict__ b3,
    const float* __restrict__ W4, const float* __restrict__ b4,
    const float* __restrict__ W5, const float* __restrict__ b5,
    float* __restrict__ dg, int total)
{
  __shared__ _Float16 smem[WAVES_PER_BLOCK][16 * 64];

  const int  lane   = threadIdx.x & 31;
  const int  laneLo = lane & 15;
  const bool hi     = (lane >= 16);
  const int  wv     = threadIdx.x >> 5;

  // ---- layer-1 weights in A-element order (per-lane private) ----
  float w1v[16], b1v[16];
  #pragma unroll
  for (int i = 0; i < 16; ++i) {
    int k = ((i < 8) ? 0 : 16) + (hi ? 8 : 0) + (i & 7);
    w1v[i] = W1[k];
    b1v[i] = b1[k];
  }

  // ---- weight matrices pre-packed as B operands (f16), kept in VGPRs ----
  // W stored [Nout, Kin] row-major; tile (kh, nt): n = nt*16+laneLo,
  // element i -> k = kh*32 + ((i<8)?0:16) + (hi?8:0) + (i&7)
  v16h B2[4], B3[2][4], B4[2][2];
  {
    const int khi = hi ? 8 : 0;
    #pragma unroll
    for (int nt = 0; nt < 4; ++nt) {
      int n = nt * 16 + laneLo;
      #pragma unroll
      for (int i = 0; i < 16; ++i) B2[nt][i] = (_Float16)W2[n * 32 + ((i < 8) ? 0 : 16) + khi + (i & 7)];
    }
    #pragma unroll
    for (int kh = 0; kh < 2; ++kh)
      #pragma unroll
      for (int nt = 0; nt < 4; ++nt) {
        int n = nt * 16 + laneLo;
        #pragma unroll
        for (int i = 0; i < 16; ++i)
          B3[kh][nt][i] = (_Float16)W3[n * 64 + kh * 32 + ((i < 8) ? 0 : 16) + khi + (i & 7)];
      }
    #pragma unroll
    for (int kh = 0; kh < 2; ++kh)
      #pragma unroll
      for (int nt = 0; nt < 2; ++nt) {
        int n = nt * 16 + laneLo;
        #pragma unroll
        for (int i = 0; i < 16; ++i)
          B4[kh][nt][i] = (_Float16)W4[n * 64 + kh * 32 + ((i < 8) ? 0 : 16) + khi + (i & 7)];
      }
  }
  // biases in C-layout (feature N = laneLo + nt*16, same for all 8 rows)
  float b2s[4], b3s[4], b4s[2];
  #pragma unroll
  for (int nt = 0; nt < 4; ++nt) {
    b2s[nt] = b2[laneLo + nt * 16];
    b3s[nt] = b3[laneLo + nt * 16];
  }
  b4s[0] = b4[laneLo];
  b4s[1] = b4[laneLo + 16];
  const float w5a = W5[laneLo], w5b = W5[laneLo + 16];
  const float b5v = b5[0];

  const int nTiles = total >> 4;
  const int waveId = blockIdx.x * WAVES_PER_BLOCK + wv;
  const int nWaves = gridDim.x * WAVES_PER_BLOCK;

  _Float16*       tileLds = &smem[wv][0];
  const _Float16* rowp    = tileLds + laneLo * 64;   // this lane's A-row
  const int       kb      = hi ? 8 : 0;

  for (int tile = waveId; tile < nTiles; tile += nWaves) {
    const int   base = tile * 16;
    const float tv   = t[base + laneLo];

    // ---- layer 1 (rank-1, VALU) directly in A-operand layout ----
    v16h a1;
    #pragma unroll
    for (int i = 0; i < 16; ++i)
      a1[i] = (_Float16)softplus_f(fmaf(tv, w1v[i], b1v[i]));

    // ---- layer 2: [16x32] @ [32x64] -> 4 WMMAs ----
    v8f h2[4];
    #pragma unroll
    for (int nt = 0; nt < 4; ++nt) {
      v8f z = {};
      h2[nt] = __builtin_amdgcn_wmma_f32_16x16x32_f16(
          false, a1, false, B2[nt], (short)0, z, false, false);
    }
    #pragma unroll
    for (int nt = 0; nt < 4; ++nt)
      #pragma unroll
      for (int r = 0; r < 8; ++r) {
        int M = r + (hi ? 8 : 0);
        tileLds[M * 64 + laneLo + nt * 16] =
            (_Float16)softplus_f(h2[nt][r] + b2s[nt]);
      }
    asm volatile("s_wait_dscnt 0" ::: "memory");

    // re-gather C-layout -> A-layout (16x64 = two 16x32 operands)
    v16h Alo, Ahi;
    #pragma unroll
    for (int i = 0; i < 8; ++i) {
      Alo[i]     = rowp[kb + i];
      Alo[i + 8] = rowp[16 + kb + i];
      Ahi[i]     = rowp[32 + kb + i];
      Ahi[i + 8] = rowp[48 + kb + i];
    }
    asm volatile("s_wait_dscnt 0" ::: "memory");

    // ---- layer 3: [16x64] @ [64x64] -> 8 WMMAs ----
    v8f h3[4];
    #pragma unroll
    for (int nt = 0; nt < 4; ++nt) {
      v8f z = {};
      v8f acc = __builtin_amdgcn_wmma_f32_16x16x32_f16(
          false, Alo, false, B3[0][nt], (short)0, z, false, false);
      h3[nt] = __builtin_amdgcn_wmma_f32_16x16x32_f16(
          false, Ahi, false, B3[1][nt], (short)0, acc, false, false);
    }
    #pragma unroll
    for (int nt = 0; nt < 4; ++nt)
      #pragma unroll
      for (int r = 0; r < 8; ++r) {
        int M = r + (hi ? 8 : 0);
        tileLds[M * 64 + laneLo + nt * 16] =
            (_Float16)softplus_f(h3[nt][r] + b3s[nt]);
      }
    asm volatile("s_wait_dscnt 0" ::: "memory");

    #pragma unroll
    for (int i = 0; i < 8; ++i) {
      Alo[i]     = rowp[kb + i];
      Alo[i + 8] = rowp[16 + kb + i];
      Ahi[i]     = rowp[32 + kb + i];
      Ahi[i + 8] = rowp[48 + kb + i];
    }
    asm volatile("s_wait_dscnt 0" ::: "memory");

    // ---- layer 4: [16x64] @ [64x32] -> 4 WMMAs ----
    float h4[2][8];
    #pragma unroll
    for (int nt = 0; nt < 2; ++nt) {
      v8f z = {};
      v8f acc = __builtin_amdgcn_wmma_f32_16x16x32_f16(
          false, Alo, false, B4[0][nt], (short)0, z, false, false);
      acc = __builtin_amdgcn_wmma_f32_16x16x32_f16(
          false, Ahi, false, B4[1][nt], (short)0, acc, false, false);
      #pragma unroll
      for (int r = 0; r < 8; ++r) h4[nt][r] = softplus_f(acc[r] + b4s[nt]);
    }

    // ---- layer 5: dot over 32 features; 16-lane xor reduction ----
    #pragma unroll
    for (int r = 0; r < 8; ++r) {
      float part = h4[0][r] * w5a + h4[1][r] * w5b;
      part += __shfl_xor(part, 1, 32);
      part += __shfl_xor(part, 2, 32);
      part += __shfl_xor(part, 4, 32);
      part += __shfl_xor(part, 8, 32);
      float v = softplus_f(part + b5v) + 1.0f;
      if (laneLo == r) dg[base + r + (hi ? 8 : 0)] = v;
    }
  }
}

// ---------------------------------------------------------------------------
// Kernel 2: per-row cumulative trapezoid. One block per row, 64 elems/thread
// kept fully in registers; in-place on g (holds dg on entry).
// ---------------------------------------------------------------------------
#define SCAN_THREADS 256

__global__ __launch_bounds__(SCAN_THREADS) void trapz_scan_kernel(
    const float* __restrict__ t, float* __restrict__ g, int N)
{
  __shared__ float totals[SCAN_THREADS];
  const int row   = blockIdx.x;
  const int tid   = threadIdx.x;
  const int chunk = N / SCAN_THREADS;            // 64
  const int k0    = tid * chunk;                 // row-local start index
  const long long base = (long long)row * N + k0;

  float inc[64];
  float pd = 0.0f, pt = 0.0f;
  if (k0 > 0) { pd = g[base - 1]; pt = t[base - 1]; }
  float run = 0.0f;
  #pragma unroll
  for (int j = 0; j < 64; ++j) {
    float d  = g[base + j];
    float tv = t[base + j];
    float ic = (k0 + j == 0) ? 0.0f : 0.5f * (d + pd) * (tv - pt);
    run += ic;
    inc[j] = run;
    pd = d; pt = tv;
  }

  totals[tid] = run;
  __syncthreads();               // also guarantees all pre-scan reads done
  #pragma unroll
  for (int off = 1; off < SCAN_THREADS; off <<= 1) {
    float v = (tid >= off) ? totals[tid - off] : 0.0f;
    __syncthreads();
    totals[tid] += v;
    __syncthreads();
  }
  const float offset = totals[tid] - run;        // exclusive prefix
  #pragma unroll
  for (int j = 0; j < 64; ++j)
    g[base + j] = offset + inc[j];
}

// ---------------------------------------------------------------------------
extern "C" void kernel_launch(void* const* d_in, const int* in_sizes, int n_in,
                              void* d_out, int out_size, void* d_ws, size_t ws_size,
                              hipStream_t stream) {
  const float* t  = (const float*)d_in[0];
  const float* W1 = (const float*)d_in[1];
  const float* b1 = (const float*)d_in[2];
  const float* W2 = (const float*)d_in[3];
  const float* b2 = (const float*)d_in[4];
  const float* W3 = (const float*)d_in[5];
  const float* b3 = (const float*)d_in[6];
  const float* W4 = (const float*)d_in[7];
  const float* b4 = (const float*)d_in[8];
  const float* W5 = (const float*)d_in[9];
  const float* b5 = (const float*)d_in[10];
  float* out = (float*)d_out;

  const int total = in_sizes[0];        // B*N = 4,194,304
  const int B = 256;
  const int N = total / B;              // 16384

  (void)d_ws; (void)ws_size; (void)n_in; (void)out_size;

  // 1024 blocks x 8 waves = 8192 waves; 262144 tiles -> 32 tiles/wave,
  // amortizing the one-time weight register packing.
  mlp_wmma_kernel<<<1024, 256, 0, stream>>>(
      t, W1, b1, W2, b2, W3, b3, W4, b4, W5, b5, out, total);

  trapz_scan_kernel<<<B, SCAN_THREADS, 0, stream>>>(t, out, N);
}
